// HypergraphEdgeBlock_28286654612013
// MI455X (gfx1250) — matrix operand order repaired and checked by reading, not compiled, not run on hardware
//
#include <hip/hip_runtime.h>
#include <hip/hip_bf16.h>

typedef __attribute__((ext_vector_type(16))) __bf16 v16bf;
typedef __attribute__((ext_vector_type(8)))  float  v8f;
typedef __attribute__((ext_vector_type(4)))  float  v4f;
typedef __attribute__((ext_vector_type(4)))  __bf16 v4bf;

#define D_EDGE 256
#define D_NODE 256
#define D_GLOB 128
#define KNB    4
#define D_IN   1408          // 256 + 4*256 + 128
#define H1     512
#define H2     512
#define H3     256
#define NK1    (D_IN/32)     // 44
#define NK2    (H1/32)       // 16
#define NK3    (H2/32)       // 16
#define TM     16            // edges per workgroup
#define XSTRIDE 1416         // padded bf16 row stride (bank-conflict-free b128 reads)
#define HSTRIDE 520          // padded bf16 row stride for h1/h2

#define R0_BYTES (TM*XSTRIDE*2)            // 45312 : X, later h2, later LN scratch
#define R1_OFF   (R0_BYTES)
#define R1_BYTES (TM*HSTRIDE*2)            // 16640 : h1, later h3 (f32 16*256*4=16384)
#define IDX_OFF  (R0_BYTES + R1_BYTES)     // 61952

// ---------------- prep kernels ----------------

__global__ void prefix_kernel(const int* __restrict__ num, int* __restrict__ offs, int G) {
  if (blockIdx.x == 0 && threadIdx.x == 0) {
    int a = 0; offs[0] = 0;
    for (int g = 0; g < G; ++g) { a += num[g]; offs[g + 1] = a; }
  }
}

__global__ void gid_fill_kernel(const int* __restrict__ offs, int* __restrict__ gid, int E, int G) {
  int e = blockIdx.x * blockDim.x + threadIdx.x;
  if (e >= E) return;
  int lo = 0, hi = G;
  while (hi - lo > 1) { int mid = (lo + hi) >> 1; if (e >= offs[mid]) lo = mid; else hi = mid; }
  gid[e] = lo;
}

// Pack f32 [K,N] weight into bf16 WMMA B-fragment order:
// tile t = ntile*nk + ktile ; per lane 16 contiguous bf16 (32B) = the K-rows this
// lane holds for column (ntile*16 + lane&15); lanes 16..31 carry K offset +16.
__global__ void pack_w_kernel(const float* __restrict__ W, unsigned short* __restrict__ dst,
                              int nk, int N, int total) {
  int tid = blockIdx.x * blockDim.x + threadIdx.x;
  if (tid >= total) return;
  int lane  = tid & 31;
  int t     = tid >> 5;
  int ktile = t % nk;
  int ntile = t / nk;
  int col   = ntile * 16 + (lane & 15);
  int kbase = ktile * 32 + ((lane >> 4) << 4);
  unsigned short* d = dst + ((size_t)t * 32 + lane) * 16;
  const float* s = W + (size_t)kbase * N + col;
#pragma unroll
  for (int i = 0; i < 16; ++i) {
    union { __bf16 b; unsigned short u; } cv;
    cv.b = (__bf16)s[(size_t)i * N];
    d[i] = cv.u;
  }
}

// ---------------- fused MLP kernel ----------------

__device__ inline v16bf load_afrag(const __bf16* base, int stride, int lane, int kbase) {
  // A fragment in this code's contiguous k order (same as packed B, not AMD's table order): lane m (0-15) row m, K 0..15; lanes 16-31: K 16..31
  const __bf16* p = base + (lane & 15) * stride + kbase + ((lane >> 4) << 4);
  v16bf a;
  ((v4f*)&a)[0] = *(const v4f*)(p);        // 8 bf16 : K +0..7
  ((v4f*)&a)[1] = *(const v4f*)(p + 8);    // 8 bf16 : K +8..15
  return a;
}

__device__ inline v16bf load_bfrag(const __bf16* pw, size_t tileIdx, int lane) {
  const __bf16* p = pw + (tileIdx * 32 + lane) * 16;
  v16bf b;
  ((v4f*)&b)[0] = *(const v4f*)(p);
  ((v4f*)&b)[1] = *(const v4f*)(p + 8);
  return b;
}

__global__ __launch_bounds__(256) void fused_mlp_kernel(
    const float* __restrict__ feat, const float* __restrict__ nodes,
    const float* __restrict__ gfeat, const int* __restrict__ ind,
    const int* __restrict__ gid,
    const unsigned short* __restrict__ pw1_u, const unsigned short* __restrict__ pw2_u,
    const unsigned short* __restrict__ pw3_u,
    const float* __restrict__ b1, const float* __restrict__ b2, const float* __restrict__ b3,
    const float* __restrict__ gamma, const float* __restrict__ beta,
    float* __restrict__ out)
{
  __shared__ __align__(16) unsigned char smem[IDX_OFF + 256 + 64];
  __bf16* Xs   = (__bf16*)smem;                 // region0: X [16][XSTRIDE] bf16
  __bf16* h2s  = (__bf16*)smem;                 // region0 reuse: h2 [16][HSTRIDE] bf16
  float*  sums = (float*)smem;                  // region0 reuse: LN partial sums [256]
  float*  ssum = (float*)(smem + 1024);         // LN partial sumsq [256]
  float*  muA  = (float*)(smem + 2048);         // [16]
  float*  rsA  = (float*)(smem + 2048 + 64);    // [16]
  __bf16* h1s  = (__bf16*)(smem + R1_OFF);      // region1: h1 [16][HSTRIDE] bf16
  float*  h3s  = (float*)(smem + R1_OFF);       // region1 reuse: h3 [16][256] f32
  int*    ndidx  = (int*)(smem + IDX_OFF);      // [16][4]
  int*    gidArr = (int*)(smem + IDX_OFF + 256);// [16]

  const __bf16* pw1 = (const __bf16*)pw1_u;
  const __bf16* pw2 = (const __bf16*)pw2_u;
  const __bf16* pw3 = (const __bf16*)pw3_u;

  const int t    = threadIdx.x;
  const int lane = t & 31;
  const int wave = t >> 5;
  const int e0   = blockIdx.x * TM;

  if (t < TM * KNB)                ndidx[t] = ind[(size_t)e0 * KNB + t];
  else if (t < TM * KNB + TM)      gidArr[t - TM * KNB] = gid[e0 + (t - TM * KNB)];
  __syncthreads();

  // ---- build X tile (gather + concat + f32->bf16) ----
  for (int idx = t; idx < TM * (D_IN / 4); idx += 256) {
    int r   = idx / (D_IN / 4);
    int c4  = idx - r * (D_IN / 4);
    int col = c4 * 4;
    v4f v;
    if (col < D_EDGE) {
      v = *(const v4f*)(feat + (size_t)(e0 + r) * D_EDGE + col);
    } else if (col < D_EDGE + KNB * D_NODE) {
      int j  = (col - D_EDGE) >> 8;
      int nc = (col - D_EDGE) & 255;
      v = *(const v4f*)(nodes + (size_t)ndidx[r * KNB + j] * D_NODE + nc);
    } else {
      v = *(const v4f*)(gfeat + (size_t)gidArr[r] * D_GLOB + (col - D_EDGE - KNB * D_NODE));
    }
    v4bf pk = { (__bf16)v.x, (__bf16)v.y, (__bf16)v.z, (__bf16)v.w };
    *(v4bf*)(Xs + r * XSTRIDE + col) = pk;
  }
  __syncthreads();

  const int mhi = (lane >> 4) << 3;
  const int nl  = lane & 15;

  // ---- stage 1: [16,1408] x [1408,512] -> relu -> h1 bf16 ----
  {
    v8f acc[4] = {};
    for (int kt = 0; kt < NK1; ++kt) {
      v16bf a = load_afrag(Xs, XSTRIDE, lane, kt * 32);
      // branch-free speculative prefetch of next k-step's B fragments
      __builtin_prefetch(pw1 + ((size_t)((wave * 4) * NK1 + kt + 1) * 32 + lane) * 16, 0, 0);
      v16bf b[4];
#pragma unroll
      for (int nt = 0; nt < 4; ++nt)
        b[nt] = load_bfrag(pw1, (size_t)(wave * 4 + nt) * NK1 + kt, lane);
#pragma unroll
      for (int nt = 0; nt < 4; ++nt)
        acc[nt] = __builtin_amdgcn_wmma_f32_16x16x32_bf16(
            false, a, false, b[nt], (short)0, acc[nt], false, false);
    }
#pragma unroll
    for (int nt = 0; nt < 4; ++nt) {
      int colbase = (wave * 4 + nt) * 16;
      float bias = b1[colbase + nl];
#pragma unroll
      for (int v = 0; v < 8; ++v) {
        float x = acc[nt][v] + bias;
        h1s[(v + mhi) * HSTRIDE + colbase + nl] = (__bf16)(x > 0.f ? x : 0.f);
      }
    }
  }
  __syncthreads();

  // ---- stage 2: [16,512] x [512,512] -> relu -> h2 bf16 (region0) ----
  {
    v8f acc[4] = {};
    for (int kt = 0; kt < NK2; ++kt) {
      v16bf a = load_afrag(h1s, HSTRIDE, lane, kt * 32);
      v16bf b[4];
#pragma unroll
      for (int nt = 0; nt < 4; ++nt)
        b[nt] = load_bfrag(pw2, (size_t)(wave * 4 + nt) * NK2 + kt, lane);
#pragma unroll
      for (int nt = 0; nt < 4; ++nt)
        acc[nt] = __builtin_amdgcn_wmma_f32_16x16x32_bf16(
            false, a, false, b[nt], (short)0, acc[nt], false, false);
    }
#pragma unroll
    for (int nt = 0; nt < 4; ++nt) {
      int colbase = (wave * 4 + nt) * 16;
      float bias = b2[colbase + nl];
#pragma unroll
      for (int v = 0; v < 8; ++v) {
        float x = acc[nt][v] + bias;
        h2s[(v + mhi) * HSTRIDE + colbase + nl] = (__bf16)(x > 0.f ? x : 0.f);
      }
    }
  }
  __syncthreads();

  // ---- stage 3: [16,512] x [512,256] -> sigmoid -> h3 f32 (region1) ----
  {
    v8f acc[2] = {};
    for (int kt = 0; kt < NK3; ++kt) {
      v16bf a = load_afrag(h2s, HSTRIDE, lane, kt * 32);
      v16bf b[2];
#pragma unroll
      for (int nt = 0; nt < 2; ++nt)
        b[nt] = load_bfrag(pw3, (size_t)(wave * 2 + nt) * NK3 + kt, lane);
#pragma unroll
      for (int nt = 0; nt < 2; ++nt)
        acc[nt] = __builtin_amdgcn_wmma_f32_16x16x32_bf16(
            false, a, false, b[nt], (short)0, acc[nt], false, false);
    }
#pragma unroll
    for (int nt = 0; nt < 2; ++nt) {
      int colbase = (wave * 2 + nt) * 16;
      float bias = b3[colbase + nl];
#pragma unroll
      for (int v = 0; v < 8; ++v) {
        float x = acc[nt][v] + bias;
        h3s[(v + mhi) * H3 + colbase + nl] = 1.0f / (1.0f + __expf(-x));
      }
    }
  }
  __syncthreads();

  // ---- LayerNorm over 256 cols per row (16 threads per row) ----
  {
    int r = t >> 4, seg = t & 15;
    const float* hr = h3s + r * H3 + seg * 16;
    float s = 0.f, q = 0.f;
#pragma unroll
    for (int i = 0; i < 16; ++i) { float x = hr[i]; s += x; q += x * x; }
    sums[t] = s; ssum[t] = q;
  }
  __syncthreads();
  if ((t & 15) == 0) {
    int r = t >> 4;
    float s = 0.f, q = 0.f;
    for (int j = 0; j < 16; ++j) { s += sums[r * 16 + j]; q += ssum[r * 16 + j]; }
    float mu  = s * (1.0f / 256.0f);
    float var = q * (1.0f / 256.0f) - mu * mu;
    muA[r] = mu;
    rsA[r] = rsqrtf(var + 1e-3f);
  }
  __syncthreads();
  {
    int r = t >> 4, seg = t & 15;
    float mu = muA[r], rstd = rsA[r];
    const float* hr = h3s + r * H3 + seg * 16;
    float* op = out + (size_t)(e0 + r) * H3 + seg * 16;
#pragma unroll
    for (int i = 0; i < 4; ++i) {
      v4f h  = *(const v4f*)(hr + i * 4);
      v4f g  = *(const v4f*)(gamma + seg * 16 + i * 4);
      v4f bt = *(const v4f*)(beta + seg * 16 + i * 4);
      v4f o  = (h - mu) * rstd * g + bt;
      *(v4f*)(op + i * 4) = o;
    }
  }
}

// ---------------- host launcher ----------------

extern "C" void kernel_launch(void* const* d_in, const int* in_sizes, int n_in,
                              void* d_out, int out_size, void* d_ws, size_t ws_size,
                              hipStream_t stream) {
  (void)n_in; (void)out_size; (void)ws_size;
  const float* feat  = (const float*)d_in[0];
  const float* nodes = (const float*)d_in[1];
  const float* gfeat = (const float*)d_in[2];
  const int*   ind   = (const int*)d_in[3];
  const int*   num   = (const int*)d_in[4];
  const float* W1    = (const float*)d_in[5];
  const float* b1    = (const float*)d_in[6];
  const float* W2    = (const float*)d_in[7];
  const float* b2    = (const float*)d_in[8];
  const float* W3    = (const float*)d_in[9];
  const float* b3    = (const float*)d_in[10];
  const float* gamma = (const float*)d_in[11];
  const float* beta  = (const float*)d_in[12];
  float* out = (float*)d_out;

  int E = in_sizes[0] / D_EDGE;   // 102400
  int G = in_sizes[4];            // 64

  char* ws = (char*)d_ws;
  int* offs = (int*)ws;                                  // (G+1) ints
  int* gid  = (int*)(ws + 512);                          // E ints
  size_t gidBytes = (((size_t)E * 4) + 255) & ~(size_t)255;
  unsigned short* pw1 = (unsigned short*)(ws + 512 + gidBytes);
  size_t w1b = (size_t)(H1 / 16) * NK1 * 32 * 16 * 2;    // 1.44 MB
  unsigned short* pw2 = (unsigned short*)((char*)pw1 + w1b);
  size_t w2b = (size_t)(H2 / 16) * NK2 * 32 * 16 * 2;    // 0.5 MB
  unsigned short* pw3 = (unsigned short*)((char*)pw2 + w2b);

  prefix_kernel<<<1, 1, 0, stream>>>(num, offs, G);
  gid_fill_kernel<<<(E + 255) / 256, 256, 0, stream>>>(offs, gid, E, G);

  int tot1 = (H1 / 16) * NK1 * 32;
  int tot2 = (H2 / 16) * NK2 * 32;
  int tot3 = (H3 / 16) * NK3 * 32;
  pack_w_kernel<<<(tot1 + 255) / 256, 256, 0, stream>>>(W1, pw1, NK1, H1, tot1);
  pack_w_kernel<<<(tot2 + 255) / 256, 256, 0, stream>>>(W2, pw2, NK2, H2, tot2);
  pack_w_kernel<<<(tot3 + 255) / 256, 256, 0, stream>>>(W3, pw3, NK3, H3, tot3);

  fused_mlp_kernel<<<E / TM, 256, 0, stream>>>(feat, nodes, gfeat, ind, gid,
                                               pw1, pw2, pw3, b1, b2, b3,
                                               gamma, beta, out);
}
